// MoELayer_41575283425900
// MI455X (gfx1250) — compile-verified, matrix operand order checked by reading, not run on hardware
//
#include <hip/hip_runtime.h>
#include <hip/hip_bf16.h>
#include <math.h>

// ---------------- problem constants (from reference) ----------------
#define D_MODEL 1024
#define D_FF    4096
#define N_EXP   8
#define TOPK    2
#define T_TOK   8192                 // B*S = 4*2048
#define N_ASSIGN (T_TOK * TOPK)      // 16384

typedef __attribute__((ext_vector_type(16))) __bf16 v16bf;
typedef __attribute__((ext_vector_type(8)))  float  v8f;
typedef __attribute__((ext_vector_type(4)))  int    v4i;

typedef unsigned short u16;
typedef unsigned int   u32;

// ---------------- CDNA5 async global->LDS copy (ASYNCcnt path) ----------------
#if defined(__has_builtin)
#  if __has_builtin(__builtin_amdgcn_global_load_async_to_lds_b128)
#    define HAVE_ASYNC_LDS 1
#  endif
#endif
#ifndef HAVE_ASYNC_LDS
#  define HAVE_ASYNC_LDS 0
#endif

#if HAVE_ASYNC_LDS
__device__ __forceinline__ void async_cp_b128(const u16* g, u16* l) {
  // global_load_async_to_lds_b128: copy engine writes LDS, no VGPR data path.
  // Param 0: v4i in AS(1) (global src); param 1: LDS dst; imm offset; imm cpol.
  __builtin_amdgcn_global_load_async_to_lds_b128(
      (__attribute__((address_space(1))) v4i*)(v4i*)g,
      (__attribute__((address_space(3))) v4i*)(v4i*)l, 0, 0);
}
#  define WAIT_ASYNC() asm volatile("s_wait_asynccnt 0" ::: "memory")
#else
#  define WAIT_ASYNC() ((void)0)
#endif

// ---------------- helpers ----------------
__device__ __forceinline__ u16 f32_to_bf16(float f) {
  u32 b = __float_as_uint(f);
  u32 r = b + 0x7FFFu + ((b >> 16) & 1u);   // round-to-nearest-even
  return (u16)(r >> 16);
}

union FragU {
  uint4 q[2];
  v16bf v;
};

__device__ __forceinline__ v16bf load_frag_lds(const u16* p) {
  FragU f;
  f.q[0] = *reinterpret_cast<const uint4*>(p);
  f.q[1] = *reinterpret_cast<const uint4*>(p + 8);
  return f.v;
}

// ---------------- kernel 1: fp32 -> bf16 cast (grid-stride) ----------------
__global__ __launch_bounds__(256) void cast_f32_bf16(const float* __restrict__ src,
                                                     u16* __restrict__ dst, int n) {
  int i = blockIdx.x * blockDim.x + threadIdx.x;
  int stride = gridDim.x * blockDim.x;
  for (; i < n; i += stride) dst[i] = f32_to_bf16(src[i]);
}

// ---------------- kernel 2: zero the atomic counters ----------------
__global__ void init_counts(int* c1, int* c2) {
  if (threadIdx.x < N_EXP) { c1[threadIdx.x] = 0; c2[threadIdx.x] = 0; }
}

// ---------------- kernel 3: router (one wave32 per token) ----------------
__global__ __launch_bounds__(256) void router_topk(
    const float* __restrict__ x, const float* __restrict__ rw,
    const float* __restrict__ rb, int* __restrict__ counts,
    int* __restrict__ topIdx, float* __restrict__ topW) {
  int wave = threadIdx.x >> 5;
  int lane = threadIdx.x & 31;
  int t = blockIdx.x * 8 + wave;
  if (t >= T_TOK) return;

  const float* xr = x + (size_t)t * D_MODEL;
  float acc[N_EXP];
#pragma unroll
  for (int e = 0; e < N_EXP; ++e) acc[e] = 0.f;
  for (int i = lane; i < D_MODEL; i += 32) {
    float xv = xr[i];
#pragma unroll
    for (int e = 0; e < N_EXP; ++e) acc[e] = fmaf(xv, rw[e * D_MODEL + i], acc[e]);
  }
#pragma unroll
  for (int e = 0; e < N_EXP; ++e) {
#pragma unroll
    for (int off = 16; off >= 1; off >>= 1) acc[e] += __shfl_xor(acc[e], off, 32);
  }
  if (lane == 0) {
    float logit[N_EXP];
    float mx = -1e30f;
#pragma unroll
    for (int e = 0; e < N_EXP; ++e) { logit[e] = acc[e] + rb[e]; mx = fmaxf(mx, logit[e]); }
    float p[N_EXP];
#pragma unroll
    for (int e = 0; e < N_EXP; ++e) p[e] = __expf(logit[e] - mx);
    // top-2 (ratio invariant to softmax normalizer); first index wins ties like lax.top_k
    int e1 = 0;
    for (int e = 1; e < N_EXP; ++e) if (p[e] > p[e1]) e1 = e;
    int e2 = (e1 == 0) ? 1 : 0;
    for (int e = 0; e < N_EXP; ++e) if (e != e1 && p[e] > p[e2]) e2 = e;
    float inv = 1.0f / (p[e1] + p[e2]);
    topIdx[t * 2 + 0] = e1; topIdx[t * 2 + 1] = e2;
    topW[t * 2 + 0] = p[e1] * inv; topW[t * 2 + 1] = p[e2] * inv;
    atomicAdd(&counts[e1], 1);
    atomicAdd(&counts[e2], 1);
  }
}

// ---------------- kernel 4: tiny prefix scan over 8 experts ----------------
__global__ void scan_offsets(const int* __restrict__ counts, int* __restrict__ offsets) {
  if (threadIdx.x == 0) {
    int s = 0;
    for (int e = 0; e < N_EXP; ++e) { offsets[e] = s; s += counts[e]; }
    offsets[N_EXP] = s;
  }
}

// ---------------- kernel 5: build expert bins / gather lists ----------------
__global__ __launch_bounds__(256) void assign_slots(
    const int* __restrict__ topIdx, const float* __restrict__ topW,
    const int* __restrict__ offsets, int* __restrict__ cursor,
    int* __restrict__ bins, float* __restrict__ gateW, int* __restrict__ tok2slot) {
  int a = blockIdx.x * blockDim.x + threadIdx.x;  // assignment id in [0, T*K)
  if (a >= N_ASSIGN) return;
  int e = topIdx[a];
  int slot = atomicAdd(&cursor[e], 1);
  int g = offsets[e] + slot;
  bins[g] = a >> 1;        // token id
  gateW[g] = topW[a];
  tok2slot[a] = g;
}

// ---------------- WMMA GEMM: 128x128x32 tiles, double-buffered LDS ----------------
#define BM 128
#define BN 128
#define BK 32
#define ASTR 40  // padded LDS row stride (elements); rows stay 16B aligned, kills bank conflicts

__device__ __forceinline__ void compute_tile(const u16* tA, const u16* tB,
                                             v8f acc[2][4], int wm, int wn, int lane) {
  int frow = lane & 15, fkh = lane >> 4;
  v16bf af[2], bfm[4];
#pragma unroll
  for (int i = 0; i < 2; ++i)
    af[i] = load_frag_lds(&tA[(wm * 32 + i * 16 + frow) * ASTR + fkh * 16]);
#pragma unroll
  for (int j = 0; j < 4; ++j)
    bfm[j] = load_frag_lds(&tB[(wn * 64 + j * 16 + frow) * ASTR + fkh * 16]);
#pragma unroll
  for (int i = 0; i < 2; ++i)
#pragma unroll
    for (int j = 0; j < 4; ++j)
      acc[i][j] = __builtin_amdgcn_wmma_f32_16x16x32_bf16(
          false, af[i], false, bfm[j], (short)0, acc[i][j], false, false);
}

// FIRST=true : H[g,:]    = gelu( x[bins[g],:] @ w1[e]^T + b1[e] )        (bf16 out)
// FIRST=false: out2[g,:] = gate[g] * ( H[g,:] @ w2[e]^T + b2[e] )        (fp32 out)
template <int KDIM, int NDIM, bool FIRST>
__global__ __launch_bounds__(256) void moe_gemm(
    const u16* __restrict__ Abase, const u16* __restrict__ Wbase,
    const float* __restrict__ bias, const int* __restrict__ bins,
    const float* __restrict__ gateW, const int* __restrict__ offsets,
    u16* __restrict__ Hout, float* __restrict__ Fout) {
  __shared__ u16 ldsA[2][BM * ASTR];
  __shared__ u16 ldsB[2][BN * ASTR];
  __shared__ int rowSrc[BM];

  int e = blockIdx.z;
  int base = offsets[e];
  int rows = offsets[e + 1] - base;
  int tm = blockIdx.y;
  if (tm * BM >= rows) return;   // block-uniform exit (EXEC stays all-ones for WMMA)
  int tn = blockIdx.x;
  int tid = threadIdx.x;

  // A-row source indices, clamped to a valid row => branch-free, always-legal loads.
  // Rows >= `rows` compute garbage that is masked at the store.
  if (tid < BM) {
    int m = tm * BM + tid;
    if (m >= rows) m = rows - 1;
    rowSrc[tid] = FIRST ? bins[base + m] : (base + m);
  }
  __syncthreads();

  const u16* Wexp = Wbase + (size_t)e * NDIM * KDIM;

  int wv = tid >> 5, lane = tid & 31;
  int wm = wv & 3, wn = wv >> 2;   // 4x2 wave grid: each wave owns 32(M) x 64(N)

  v8f acc[2][4];
  v8f zero = {0, 0, 0, 0, 0, 0, 0, 0};
#pragma unroll
  for (int i = 0; i < 2; ++i)
#pragma unroll
    for (int j = 0; j < 4; ++j) acc[i][j] = zero;

  int ldRow = tid >> 1;     // 0..127
  int ldHalf = tid & 1;     // which 16-element half of the 32-element K slice

  auto issue = [&](int buf, int k0) {
    int ar = rowSrc[ldRow];
    const u16* ga = Abase + (size_t)ar * KDIM + k0 + ldHalf * 16;
    const u16* gb = Wexp + (size_t)(tn * BN + ldRow) * KDIM + k0 + ldHalf * 16;
    __builtin_prefetch(gb + BK, 0, 1);       // global_prefetch_b8 for the K-step after next
    u16* sA = &ldsA[buf][ldRow * ASTR + ldHalf * 16];
    u16* sB = &ldsB[buf][ldRow * ASTR + ldHalf * 16];
#if HAVE_ASYNC_LDS
    async_cp_b128(ga, sA);
    async_cp_b128(ga + 8, sA + 8);
    async_cp_b128(gb, sB);
    async_cp_b128(gb + 8, sB + 8);
#else
    uint4 a0 = *reinterpret_cast<const uint4*>(ga);
    uint4 a1 = *reinterpret_cast<const uint4*>(ga + 8);
    uint4 b0 = *reinterpret_cast<const uint4*>(gb);
    uint4 b1v = *reinterpret_cast<const uint4*>(gb + 8);
    *reinterpret_cast<uint4*>(sA) = a0;
    *reinterpret_cast<uint4*>(sA + 8) = a1;
    *reinterpret_cast<uint4*>(sB) = b0;
    *reinterpret_cast<uint4*>(sB + 8) = b1v;
#endif
  };

  constexpr int NSTEP = KDIM / BK;
  issue(0, 0);
  for (int s = 0; s < NSTEP; ++s) {
    WAIT_ASYNC();        // stage-s copies (this wave) have landed in LDS
    __syncthreads();     // all waves' stage-s data visible; all done reading buf[(s+1)&1]
    if (s + 1 < NSTEP) issue((s + 1) & 1, (s + 1) * BK);   // overlap next copies w/ WMMA
    compute_tile(ldsA[s & 1], ldsB[s & 1], acc, wm, wn, lane);
  }

  // epilogue: C layout — VGPR r holds M = r + 8*(lane>=16); N = lane&15
  int lhalf = lane >> 4;
  int ncol = lane & 15;
#pragma unroll
  for (int i = 0; i < 2; ++i) {
#pragma unroll
    for (int j = 0; j < 4; ++j) {
      int nglob = tn * BN + wn * 64 + j * 16 + ncol;
      float b = bias[e * NDIM + nglob];
#pragma unroll
      for (int r = 0; r < 8; ++r) {
        int m = tm * BM + wm * 32 + i * 16 + lhalf * 8 + r;
        if (m < rows) {
          float v = acc[i][j][r] + b;
          if (FIRST) {
            float gel = 0.5f * v * (1.0f + erff(v * 0.70710678118654752f)); // exact gelu
            Hout[(size_t)(base + m) * NDIM + nglob] = f32_to_bf16(gel);
          } else {
            Fout[(size_t)(base + m) * NDIM + nglob] = gateW[base + m] * v;
          }
        }
      }
    }
  }
}

// ---------------- kernel 8: combine the 2 expert contributions per token ----------------
__global__ __launch_bounds__(256) void combine_out(
    const float* __restrict__ out2, const int* __restrict__ tok2slot,
    float* __restrict__ out) {
  size_t i = (size_t)blockIdx.x * blockDim.x + threadIdx.x;
  if (i >= (size_t)T_TOK * D_MODEL) return;
  int t = (int)(i / D_MODEL);
  int d = (int)(i % D_MODEL);
  int g0 = tok2slot[t * 2 + 0];
  int g1 = tok2slot[t * 2 + 1];
  out[i] = out2[(size_t)g0 * D_MODEL + d] + out2[(size_t)g1 * D_MODEL + d];
}

// ---------------- launcher ----------------
static inline size_t align_up(size_t v, size_t a) { return (v + a - 1) & ~(a - 1); }

extern "C" void kernel_launch(void* const* d_in, const int* in_sizes, int n_in,
                              void* d_out, int out_size, void* d_ws, size_t ws_size,
                              hipStream_t stream) {
  const float* x  = (const float*)d_in[0];
  const float* rw = (const float*)d_in[1];
  const float* rb = (const float*)d_in[2];
  const float* w1 = (const float*)d_in[3];
  const float* b1 = (const float*)d_in[4];
  const float* w2 = (const float*)d_in[5];
  const float* b2 = (const float*)d_in[6];
  float* out = (float*)d_out;

  // workspace carve-up (~353 MB)
  char* p = (char*)d_ws;
  size_t off = 0;
  u16* xb  = (u16*)(p + off);  off = align_up(off + (size_t)T_TOK * D_MODEL * 2, 256);
  u16* w1b = (u16*)(p + off);  off = align_up(off + (size_t)N_EXP * D_FF * D_MODEL * 2, 256);
  u16* w2b = (u16*)(p + off);  off = align_up(off + (size_t)N_EXP * D_MODEL * D_FF * 2, 256);
  u16* H   = (u16*)(p + off);  off = align_up(off + (size_t)N_ASSIGN * D_FF * 2, 256);
  float* out2 = (float*)(p + off); off = align_up(off + (size_t)N_ASSIGN * D_MODEL * 4, 256);
  int* counts  = (int*)(p + off);  off = align_up(off + 64, 256);
  int* cursor  = (int*)(p + off);  off = align_up(off + 64, 256);
  int* offsets = (int*)(p + off);  off = align_up(off + (N_EXP + 1) * 4, 256);
  int* bins    = (int*)(p + off);  off = align_up(off + (size_t)N_ASSIGN * 4, 256);
  float* gateW = (float*)(p + off); off = align_up(off + (size_t)N_ASSIGN * 4, 256);
  int* topIdx  = (int*)(p + off);  off = align_up(off + (size_t)N_ASSIGN * 4, 256);
  float* topW  = (float*)(p + off); off = align_up(off + (size_t)N_ASSIGN * 4, 256);
  int* tok2slot = (int*)(p + off); off = align_up(off + (size_t)N_ASSIGN * 4, 256);
  (void)ws_size; (void)n_in; (void)in_sizes; (void)out_size;

  // 1) cast activations + weights to bf16
  cast_f32_bf16<<<4096, 256, 0, stream>>>(x,  xb,  T_TOK * D_MODEL);
  cast_f32_bf16<<<4096, 256, 0, stream>>>(w1, w1b, N_EXP * D_FF * D_MODEL);
  cast_f32_bf16<<<4096, 256, 0, stream>>>(w2, w2b, N_EXP * D_MODEL * D_FF);

  // 2) routing
  init_counts<<<1, 32, 0, stream>>>(counts, cursor);
  router_topk<<<T_TOK / 8, 256, 0, stream>>>(x, rw, rb, counts, topIdx, topW);
  scan_offsets<<<1, 32, 0, stream>>>(counts, offsets);
  assign_slots<<<N_ASSIGN / 256, 256, 0, stream>>>(topIdx, topW, offsets, cursor,
                                                   bins, gateW, tok2slot);

  // 3) expert GEMMs (grid.y covers worst-case rows; blocks past count exit early)
  dim3 g1(D_FF / BN, (T_TOK + BM - 1) / BM, N_EXP);
  moe_gemm<D_MODEL, D_FF, true><<<g1, 256, 0, stream>>>(xb, w1b, b1, bins, nullptr,
                                                        offsets, H, nullptr);
  dim3 g2(D_MODEL / BN, (T_TOK + BM - 1) / BM, N_EXP);
  moe_gemm<D_FF, D_MODEL, false><<<g2, 256, 0, stream>>>(H, w2b, b2, nullptr, gateW,
                                                         offsets, nullptr, out2);

  // 4) combine top-2 contributions (fixed order => deterministic)
  size_t total = (size_t)T_TOK * D_MODEL;
  combine_out<<<(unsigned)((total + 255) / 256), 256, 0, stream>>>(out2, tok2slot, out);
}
